// RevRNNModel_28484223107779
// MI455X (gfx1250) — matrix operand order repaired and checked by reading, not compiled, not run on hardware
//
#include <hip/hip_runtime.h>
#include <hip/hip_bf16.h>

typedef __attribute__((ext_vector_type(16))) __bf16        v16bf;
typedef __attribute__((ext_vector_type(2)))  __bf16        v2bf;
typedef __attribute__((ext_vector_type(8)))  float         v8f;
typedef __attribute__((ext_vector_type(4)))  float         fv4;
typedef __attribute__((ext_vector_type(4)))  unsigned int  uv4;
typedef __attribute__((ext_vector_type(2)))  unsigned int  uv2;

#define NTOK  32000
#define NINP  1024
#define NHID  1024
#define BATCH 4096

#define BM 128
#define BN 128
#define BK 32
#define LSTR 48   // LDS row stride in bf16 elements (32 + 16 pad, keeps 16B alignment)

// pack two f32 -> packed bf16 pair (low = bf16(a), high = bf16(b))
__device__ __forceinline__ unsigned pack2_bf16(float a, float b) {
#if __has_builtin(__builtin_amdgcn_cvt_pk_bf16_f32)
    v2bf r = __builtin_amdgcn_cvt_pk_bf16_f32(a, b);
    return __builtin_bit_cast(unsigned, r);
#else
    // round-to-nearest (ties away) then pack both high halves with one v_perm_b32
    unsigned ua = __float_as_uint(a) + 0x8000u;
    unsigned ub = __float_as_uint(b) + 0x8000u;
    // sel bytes (LSB..MSB): ua[2], ua[3], ub[2], ub[3]  (S1 = ua -> bytes 0..3)
    return __builtin_amdgcn_perm(ub, ua, 0x07060302u);
#endif
}

// C[M,N] = concat_k(A0,A1)[M,K] * Bw[N,K]^T + bias[N]
// A0 covers K columns [0,K0), A1 covers [K0,K). All dims multiples of tiles.
__launch_bounds__(256)
__global__ void gemm_bias_kernel(const float* __restrict__ A0,
                                 const float* __restrict__ A1,
                                 int K0, int K,
                                 const float* __restrict__ Bw,
                                 const float* __restrict__ bias,
                                 float* __restrict__ C, int N)
{
    __shared__ unsigned short lA[2][BM * LSTR];
    __shared__ unsigned short lB[2][BN * LSTR];

    const int tid  = threadIdx.x;
    const int lane = tid & 31;
    const int wave = tid >> 5;
    const int lr   = lane & 15;
    const int hi   = lane >> 4;

    const int mblk = blockIdx.y * BM;
    const int nblk = blockIdx.x * BN;

    // 8 waves in a 2x4 grid; each wave owns a 64x32 output tile (4x2 WMMA tiles)
    const int wm = (wave >> 2) * 64;
    const int wn = (wave & 3)  * 32;

    // global->LDS staging: thread loads float4 col-group g of rows r0, r0+32, r0+64, r0+96
    const int g  = tid & 7;
    const int r0 = tid >> 3;

    v8f acc[4][2];
    const v8f vzero = {0.f,0.f,0.f,0.f,0.f,0.f,0.f,0.f};
#pragma unroll
    for (int i = 0; i < 4; ++i)
#pragma unroll
        for (int j = 0; j < 2; ++j) acc[i][j] = vzero;

    const int T = K / BK;
    fv4 ra[4], rb[4];

#define LOAD_REGS(KCOL)                                                           \
    do {                                                                          \
        int kcol_ = (KCOL);                                                       \
        const float* As_; int abase_, alda_;                                      \
        if (kcol_ < K0) { As_ = A0; abase_ = kcol_;      alda_ = K0;     }        \
        else            { As_ = A1; abase_ = kcol_ - K0; alda_ = K - K0; }        \
        _Pragma("unroll")                                                         \
        for (int i_ = 0; i_ < 4; ++i_) {                                          \
            int row_ = r0 + i_ * 32;                                              \
            ra[i_] = *(const fv4*)&As_[(size_t)(mblk + row_) * alda_ + abase_ + g * 4]; \
            rb[i_] = *(const fv4*)&Bw[(size_t)(nblk + row_) * K + kcol_ + g * 4]; \
        }                                                                         \
    } while (0)

#define STORE_STAGE(BUF)                                                          \
    do {                                                                          \
        _Pragma("unroll")                                                         \
        for (int i_ = 0; i_ < 4; ++i_) {                                          \
            int row_ = r0 + i_ * 32;                                              \
            uv2 pa_, pb_;                                                         \
            pa_[0] = pack2_bf16(ra[i_][0], ra[i_][1]);                            \
            pa_[1] = pack2_bf16(ra[i_][2], ra[i_][3]);                            \
            pb_[0] = pack2_bf16(rb[i_][0], rb[i_][1]);                            \
            pb_[1] = pack2_bf16(rb[i_][2], rb[i_][3]);                            \
            *(uv2*)&lA[BUF][row_ * LSTR + g * 4] = pa_;                           \
            *(uv2*)&lB[BUF][row_ * LSTR + g * 4] = pb_;                           \
        }                                                                         \
    } while (0)

    // prologue: fill buffer 0, start fetch of k-tile 1
    LOAD_REGS(0);
    STORE_STAGE(0);
    if (T > 1) LOAD_REGS(BK);
    __syncthreads();

    union Frag { v16bf v; uv4 u[2]; };

    for (int kt = 0; kt < T; ++kt) {
        const int cur = kt & 1;

        // load fragments per ISA VGPR layouts
        Frag af[4], bfm[2];
#pragma unroll
        for (int mt = 0; mt < 4; ++mt) {
            const unsigned short* p = &lA[cur][(wm + mt * 16 + lr) * LSTR];
            af[mt].u[0] = *(const uv4*)(p + hi * 8);        // K = hi*8 .. hi*8+7
            af[mt].u[1] = *(const uv4*)(p + 16 + hi * 8);   // K = 16+hi*8 ..
        }
#pragma unroll
        for (int nt = 0; nt < 2; ++nt) {
            const unsigned short* p = &lB[cur][(wn + nt * 16 + lr) * LSTR];
            bfm[nt].u[0] = *(const uv4*)(p + hi * 8);       // K = hi*8 .. hi*8+7
            bfm[nt].u[1] = *(const uv4*)(p + 16 + hi * 8);  // K = 16+hi*8 ..
        }

#pragma unroll
        for (int mt = 0; mt < 4; ++mt)
#pragma unroll
            for (int nt = 0; nt < 2; ++nt)
                acc[mt][nt] = __builtin_amdgcn_wmma_f32_16x16x32_bf16(
                    false, af[mt].v, false, bfm[nt].v,
                    (short)0, acc[mt][nt], false, false);

        if (kt + 1 < T) {
            // regs currently hold k-tile kt+1: stage into the other buffer,
            // then start the global fetch of k-tile kt+2
            STORE_STAGE(cur ^ 1);
            if (kt + 2 < T) LOAD_REGS((kt + 2) * BK);
            __syncthreads();   // next buffer ready AND everyone done reading cur
        }
    }

#undef LOAD_REGS
#undef STORE_STAGE

    // epilogue: D layout -> row = j + 8*hi, col = lr within 16x16 tile
#pragma unroll
    for (int mt = 0; mt < 4; ++mt) {
#pragma unroll
        for (int nt = 0; nt < 2; ++nt) {
            int col = nblk + wn + nt * 16 + lr;
            float bv = bias ? bias[col] : 0.f;
#pragma unroll
            for (int j = 0; j < 8; ++j) {
                int row = mblk + wm + mt * 16 + j + 8 * hi;
                C[(size_t)row * N + col] = acc[mt][nt][j] + bv;
            }
        }
    }
}

__global__ void gather_kernel(const int* __restrict__ ids,
                              const float* __restrict__ emb,
                              float* __restrict__ out)
{
    int i  = blockIdx.x * blockDim.x + threadIdx.x;  // float4 index
    int b  = i >> 8;                                  // NINP/4 = 256 groups/row
    int cg = i & 255;
    fv4 v = *(const fv4*)&emb[(size_t)ids[b] * NINP + cg * 4];
    *(fv4*)&out[(size_t)b * NINP + cg * 4] = v;
}

__device__ __forceinline__ float sigm(float x) { return 1.f / (1.f + __expf(-x)); }

// new_c = (sigmoid(z0)*c + sigmoid(z1)*tanh(z2)) * 0.5
__global__ void cell_c_kernel(const float* __restrict__ z, const float* __restrict__ c,
                              float* __restrict__ out)
{
    int i = blockIdx.x * blockDim.x + threadIdx.x;   // BATCH*NHID threads
    int b = i >> 10;
    int h = i & (NHID - 1);
    const float* zr = z + (size_t)b * 3 * NHID;
    float f  = sigm(zr[h]);
    float ig = sigm(zr[NHID + h]);
    float cc = tanhf(zr[2 * NHID + h]);
    out[i] = (f * c[i] + ig * cc) * 0.5f;
}

// new_h = (sigmoid(z1)*h + sigmoid(z0)*tanh(z2)) * 0.5
__global__ void cell_h_kernel(const float* __restrict__ z, const float* __restrict__ hprev,
                              float* __restrict__ out)
{
    int i = blockIdx.x * blockDim.x + threadIdx.x;
    int b = i >> 10;
    int h = i & (NHID - 1);
    const float* zr = z + (size_t)b * 3 * NHID;
    float u  = sigm(zr[h]);
    float r  = sigm(zr[NHID + h]);
    float hh = tanhf(zr[2 * NHID + h]);
    out[i] = (r * hprev[i] + u * hh) * 0.5f;
}

extern "C" void kernel_launch(void* const* d_in, const int* in_sizes, int n_in,
                              void* d_out, int out_size, void* d_ws, size_t ws_size,
                              hipStream_t stream)
{
    (void)in_sizes; (void)n_in; (void)out_size; (void)ws_size;

    const int*   ids  = (const int*)  d_in[0];
    const float* c1   = (const float*)d_in[1];
    const float* h1   = (const float*)d_in[2];
    const float* c2   = (const float*)d_in[3];
    const float* h2   = (const float*)d_in[4];
    const float* embW = (const float*)d_in[5];
    const float* W1   = (const float*)d_in[6];
    const float* b1   = (const float*)d_in[7];
    const float* W2   = (const float*)d_in[8];
    const float* b2   = (const float*)d_in[9];
    const float* decW = (const float*)d_in[10];
    const float* decb = (const float*)d_in[11];

    float* out     = (float*)d_out;
    float* decoded = out;
    float* c1n     = decoded + (size_t)BATCH * NTOK;
    float* h1n     = c1n + (size_t)BATCH * NHID;
    float* c2n     = h1n + (size_t)BATCH * NHID;
    float* h2n     = c2n + (size_t)BATCH * NHID;

    float* emb = (float*)d_ws;                       // BATCH*NINP
    float* z   = emb + (size_t)BATCH * NINP;         // BATCH*3*NHID

    const dim3 blk(256);
    const dim3 ggate(3 * NHID / BN, BATCH / BM);     // 24 x 32
    const dim3 gdec(NTOK / BN, BATCH / BM);          // 250 x 32
    const int  nelem_blocks = BATCH * NHID / 256;

    const size_t gateW = (size_t)3 * NHID * (NINP + NHID); // offset of W[3:] for a cell

    gather_kernel<<<BATCH * (NINP / 4) / 256, blk, 0, stream>>>(ids, embW, emb);

    // ---- cell 1: x = emb, K = NINP + NHID ----
    gemm_bias_kernel<<<ggate, blk, 0, stream>>>(emb, h1, NINP, NINP + NHID,
                                                W1, b1, z, 3 * NHID);
    cell_c_kernel<<<nelem_blocks, blk, 0, stream>>>(z, c1, c1n);
    gemm_bias_kernel<<<ggate, blk, 0, stream>>>(emb, c1n, NINP, NINP + NHID,
                                                W1 + gateW, b1 + 3 * NHID, z, 3 * NHID);
    cell_h_kernel<<<nelem_blocks, blk, 0, stream>>>(z, h1, h1n);

    // ---- cell 2: x = h1n, K = 2*NHID ----
    gemm_bias_kernel<<<ggate, blk, 0, stream>>>(h1n, h2, NHID, 2 * NHID,
                                                W2, b2, z, 3 * NHID);
    cell_c_kernel<<<nelem_blocks, blk, 0, stream>>>(z, c2, c2n);
    gemm_bias_kernel<<<ggate, blk, 0, stream>>>(h1n, c2n, NHID, 2 * NHID,
                                                W2 + gateW, b2 + 3 * NHID, z, 3 * NHID);
    cell_h_kernel<<<nelem_blocks, blk, 0, stream>>>(z, h2, h2n);

    // ---- decoder: [4096,1024] x [1024,32000] ----
    gemm_bias_kernel<<<gdec, blk, 0, stream>>>(h2n, nullptr, NHID, NHID,
                                               decW, decb, decoded, NTOK);
}
